// VQEmbedding_67894843015555
// MI455X (gfx1250) — matrix operand order repaired and checked by reading, not compiled, not run on hardware
//
#include <hip/hip_runtime.h>
#include <hip/hip_bf16.h>

typedef __attribute__((ext_vector_type(16))) _Float16 v16h;
typedef __attribute__((ext_vector_type(8)))  float    v8f;

#define K_CODES 1024
#define DIM     64
#define B_      16
#define T_      8192
#define NROWS   (B_ * T_)        // 131072
#define NTILES  (NROWS / 32)     // 4096 tiles of 32 rows

// ---------------------------------------------------------------------------
// Phase 0: convert codebook to f16 + ||e||^2
// ---------------------------------------------------------------------------
__global__ void vq_prep(const float* __restrict__ weight,
                        _Float16* __restrict__ wf16,
                        float* __restrict__ esq) {
    int k = blockIdx.x * blockDim.x + threadIdx.x;
    if (k >= K_CODES) return;
    const float* wr = weight + (size_t)k * DIM;
    float s = 0.f;
#pragma unroll
    for (int d = 0; d < DIM; ++d) {
        float v = wr[d];
        s = fmaf(v, v, s);
        wf16[(size_t)k * DIM + d] = (_Float16)v;
    }
    esq[k] = s;
}

// ---------------------------------------------------------------------------
// Main: one wave per 32-row tile (two 16-row sets sharing B fragments).
// kt-loop unrolled x2 with ping-pong B buffers: prefetch of tile kt+1 is in
// flight during the 4 WMMAs of tile kt, and the buffer rotation is free.
// ---------------------------------------------------------------------------
__global__ void __launch_bounds__(256) vq_main(
    const float* __restrict__ inputs,      // [B, D, T] f32
    const float* __restrict__ weight,      // [K, D]    f32
    const _Float16* __restrict__ wf16,     // [K, D]    f16
    const float* __restrict__ esq,         // [K]
    float* __restrict__ outq,              // [B, D, T]
    float* __restrict__ outidx,            // [B, T] (as float values)
    float* __restrict__ partials)          // [NTILES]
{
    const int lane = threadIdx.x & 31;
    const int wave = threadIdx.x >> 5;
    const int tile = blockIdx.x * (blockDim.x >> 5) + wave;

    const int row0 = tile << 5;            // 32 rows; same b (8192 % 32 == 0)
    const int b    = row0 >> 13;
    const int t0   = row0 & (T_ - 1);
    const int lrow = lane & 15;

    const float* inb = inputs + (size_t)b * DIM * T_;

    // ---- A fragments (16-bit A 16x32 layout) for both row sets ----
    v16h a0_0, a1_0, a0_1, a1_1;
    const int hi8 = (lane >> 4) << 4;
#pragma unroll
    for (int j = 0; j < 16; ++j) {
        int d = hi8 + j;
        a0_0[j] = (_Float16)inb[(size_t)d * T_ + t0 + lrow];
        a1_0[j] = (_Float16)inb[(size_t)(d + 32) * T_ + t0 + lrow];
        a0_1[j] = (_Float16)inb[(size_t)d * T_ + t0 + 16 + lrow];
        a1_1[j] = (_Float16)inb[(size_t)(d + 32) * T_ + t0 + 16 + lrow];
    }

    float mn0[8], mn1[8];
    int   mi0[8], mi1[8];
#pragma unroll
    for (int r = 0; r < 8; ++r) {
        mn0[r] = 3.4e38f; mi0[r] = 0;
        mn1[r] = 3.4e38f; mi1[r] = 0;
    }

    const int dbase = (lane >> 4) << 4;    // B frag: lanes 16-31 hold K=16..31

    auto loadB = [&](int kt, v16h& B0, v16h& B1, float& eb) {
        const int c = (kt << 4) + lrow;
        const _Float16* w = wf16 + (size_t)c * DIM + dbase;
        B0 = *(const v16h*)(w);
        B1 = *(const v16h*)(w + 32);
        eb = esq[c];
    };

    auto compute_tile = [&](int kt, const v16h& B0, const v16h& B1, float eb) {
        const int c = (kt << 4) + lrow;
        v8f acc0 = {}, acc1 = {};
        acc0 = __builtin_amdgcn_wmma_f32_16x16x32_f16(false, a0_0, false, B0,
                                                      (short)0, acc0, false, false);
        acc0 = __builtin_amdgcn_wmma_f32_16x16x32_f16(false, a1_0, false, B1,
                                                      (short)0, acc0, false, false);
        acc1 = __builtin_amdgcn_wmma_f32_16x16x32_f16(false, a0_1, false, B0,
                                                      (short)0, acc1, false, false);
        acc1 = __builtin_amdgcn_wmma_f32_16x16x32_f16(false, a1_1, false, B1,
                                                      (short)0, acc1, false, false);
#pragma unroll
        for (int r = 0; r < 8; ++r) {
            float d0 = fmaf(-2.f, acc0[r], eb);
            float d1 = fmaf(-2.f, acc1[r], eb);
            if (d0 < mn0[r]) { mn0[r] = d0; mi0[r] = c; }   // strict < keeps first idx
            if (d1 < mn1[r]) { mn1[r] = d1; mi1[r] = c; }
        }
    };

    // ---- pipelined main loop, ping-pong B buffers ----
    v16h B0a, B1a, B0b, B1b;
    float eba, ebb;
    loadB(0, B0a, B1a, eba);
#pragma unroll 1
    for (int kt = 0; kt < K_CODES / 16; kt += 2) {
        loadB(kt + 1, B0b, B1b, ebb);
        compute_tile(kt, B0a, B1a, eba);
        loadB((kt + 2) & 63, B0a, B1a, eba);   // clamped wrap: harmless extra load
        compute_tile(kt + 1, B0b, B1b, ebb);
    }

    // ---- epilogue per row set: argmin butterfly, index + quantized writes ----
    float lsum = 0.f;

    auto finish = [&](float (&mn)[8], int (&mi)[8], int rs) {
        // cross-lane argmin within each 16-lane half (rows r / r+8)
#pragma unroll
        for (int off = 1; off < 16; off <<= 1) {
#pragma unroll
            for (int r = 0; r < 8; ++r) {
                float om = __shfl_xor(mn[r], off);
                int   oi = __shfl_xor(mi[r], off);
                if (om < mn[r] || (om == mn[r] && oi < mi[r])) { mn[r] = om; mi[r] = oi; }
            }
        }
        const int rbase = row0 + (rs << 4);
        // indices: lane0 -> rows 0..7, lane16 -> rows 8..15
        if (lane == 0 || lane == 16) {
            int base = rbase + ((lane >> 4) << 3);
#pragma unroll
            for (int r = 0; r < 8; ++r) outidx[base + r] = (float)mi[r];
        }
        // broadcast: each lane needs index of row (lane & 15)
        const int k7 = lane & 7;
        int v = mi[0];
#pragma unroll
        for (int r = 1; r < 8; ++r) v = (k7 == r) ? mi[r] : v;
        const int srcLane = (lane & 7) | ((lane & 8) << 1);
        const int myIdx = __shfl(v, srcLane);

        // quantized write (coalesced over t, nontemporal) + loss accumulation
        const float* wrow = weight + (size_t)myIdx * DIM;
        const int d0 = (lane >> 4) * 32;   // lanes 0-15: d 0..31, 16-31: d 32..63
        const size_t baseoff = (size_t)(b * DIM + d0) * T_ + t0 + (rs << 4) + lrow;
#pragma unroll
        for (int dd = 0; dd < 32; ++dd) {
            float q = wrow[d0 + dd];
            size_t off = baseoff + (size_t)dd * T_;
            float x = __builtin_nontemporal_load(inputs + off);   // last use
            __builtin_nontemporal_store(q, outq + off);           // write-once
            float diff = q - x;
            lsum = fmaf(diff, diff, lsum);
        }
    };

    finish(mn0, mi0, 0);
    finish(mn1, mi1, 1);

#pragma unroll
    for (int off = 1; off < 32; off <<= 1) lsum += __shfl_xor(lsum, off);
    if (lane == 0) partials[tile] = lsum;
}

// ---------------------------------------------------------------------------
// Deterministic loss finalize: loss = 1.25 * sum / (N*D)
// ---------------------------------------------------------------------------
__global__ void vq_loss(const float* __restrict__ partials,
                        float* __restrict__ outloss) {
    __shared__ float red[256];
    float s = 0.f;
    for (int i = threadIdx.x; i < NTILES; i += 256) s += partials[i];
    red[threadIdx.x] = s;
    __syncthreads();
    for (int stride = 128; stride > 0; stride >>= 1) {
        if (threadIdx.x < stride) red[threadIdx.x] += red[threadIdx.x + stride];
        __syncthreads();
    }
    if (threadIdx.x == 0)
        outloss[0] = 1.25f * red[0] / (float)(NROWS * DIM);
}

// ---------------------------------------------------------------------------
extern "C" void kernel_launch(void* const* d_in, const int* in_sizes, int n_in,
                              void* d_out, int out_size, void* d_ws, size_t ws_size,
                              hipStream_t stream) {
    const float* inputs = (const float*)d_in[0];   // [16, 64, 8192] f32
    const float* weight = (const float*)d_in[1];   // [1024, 64]     f32

    char* ws = (char*)d_ws;
    _Float16* wf16  = (_Float16*)ws;                                          // 128 KB
    float*    esq   = (float*)(ws + (size_t)K_CODES * DIM * 2);               // 4 KB
    float*    parts = (float*)(ws + (size_t)K_CODES * DIM * 2 + K_CODES * 4); // 16 KB

    float* outq    = (float*)d_out;                 // [B, D, T]
    float* outloss = outq + (size_t)B_ * DIM * T_;  // scalar
    float* outidx  = outloss + 1;                   // [B, T]

    vq_prep<<<(K_CODES + 255) / 256, 256, 0, stream>>>(weight, wf16, esq);
    vq_main<<<NTILES / 8, 256, 0, stream>>>(inputs, weight, wf16, esq,
                                            outq, outidx, parts);
    vq_loss<<<1, 256, 0, stream>>>(parts, outloss);
}
